// ParallelMinGRU_62938450755683
// MI455X (gfx1250) — matrix operand order repaired and checked by reading, not compiled, not run on hardware
//
#include <hip/hip_runtime.h>
#include <hip/hip_bf16.h>

// Problem constants (from reference): B=8, T=4096, DX=DH=1024
#define BB     8
#define TT     4096
#define KK_DIM 1024
#define HH     1024
#define NCHUNK 32
#define CLEN   (TT / NCHUNK)   // 128

typedef __attribute__((ext_vector_type(16))) __bf16        v16bf;
typedef __attribute__((ext_vector_type(8)))  float         v8f;
typedef __attribute__((ext_vector_type(8)))  unsigned int  v8u;

// ---------------------------------------------------------------------------
// f32 -> packed bf16x2 (round to nearest even) -- used only in the pre-pass
// ---------------------------------------------------------------------------
__device__ __forceinline__ unsigned int pack_bf16x2(float lo, float hi) {
  unsigned int ul = __builtin_bit_cast(unsigned int, lo);
  unsigned int uh = __builtin_bit_cast(unsigned int, hi);
  ul = (ul + 0x7FFFu + ((ul >> 16) & 1u)) >> 16;
  uh = (uh + 0x7FFFu + ((uh >> 16) & 1u)) >> 16;
  return (ul & 0xFFFFu) | (uh << 16);
}

// ---------------------------------------------------------------------------
// Pre-pass: convert f32 array to bf16 (RNE). One thread per 4 elements.
// Memory-bound; conversion VALU is hidden under HBM traffic.
// ---------------------------------------------------------------------------
__global__ __launch_bounds__(256)
void cvt_bf16_kernel(const float* __restrict__ src,
                     unsigned int* __restrict__ dst, int n4) {
  const int i = blockIdx.x * blockDim.x + threadIdx.x;
  if (i >= n4) return;
  const float4 f = ((const float4*)src)[i];
  uint2 o;
  o.x = pack_bf16x2(f.x, f.y);
  o.y = pack_bf16x2(f.z, f.w);
  ((uint2*)dst)[i] = o;
}

// ---------------------------------------------------------------------------
// bf16 fragment loaders: pure b128 loads, zero per-element VALU.
// A fragment (16x32, M x K): lane L<16 holds row M=L, K = {0..7}+{16..23};
// lane L>=16 holds row M=L-16, K = {8..15}+{24..31}.
// ---------------------------------------------------------------------------
__device__ __forceinline__ v16bf load_a_frag(const unsigned short* __restrict__ rowp,
                                             int kk, int half) {
  const uint4* p = (const uint4*)(rowp + kk + half * 8);
  const uint4 q0 = p[0];   // K group 0..7 (8 bf16 = 16B)
  const uint4 q1 = p[2];   // +16 elements = +32B => K group 16..23
  v8u pk = {q0.x, q0.y, q0.z, q0.w, q1.x, q1.y, q1.z, q1.w};
  return __builtin_bit_cast(v16bf, pk);
}

// B fragment (32x16, K x N): lane L%16 selects column N; lanes 0-15 hold
// K={0..7}+{16..23}, lanes 16-31 hold K={8..15}+{24..31}, packed 2 per VGPR.
// rowp = row N of W (W is (H,K) row-major, so B(k,n) = W[n*K + k]).
__device__ __forceinline__ v16bf load_b_frag(const unsigned short* __restrict__ rowp,
                                             int kk, int half) {
  const uint4* p = (const uint4*)(rowp + kk + half * 8);
  const uint4 q0 = p[0];   // K group 0..7 (+8 for upper half)
  const uint4 q1 = p[2];   // +16 elements => K group 16..23 (+8 for upper half)
  v8u pk = {q0.x, q0.y, q0.z, q0.w, q1.x, q1.y, q1.z, q1.w};
  return __builtin_bit_cast(v16bf, pk);
}

__device__ __forceinline__ v8f wmma_bf16(v16bf a, v16bf b, v8f c) {
  // (neg_a, A, neg_b, B, c_mod, C, reuse_a, reuse_b)
  return __builtin_amdgcn_wmma_f32_16x16x32_bf16(false, a, false, b,
                                                 (short)0, c, false, false);
}

// ---------------------------------------------------------------------------
// Kernel 1: fused dual GEMM + gate epilogue (bf16 inputs, f32 accumulate).
//   coeff = 1 - sigmoid(X·Wz + bz)              -> ws
//   val   = sigmoid(X·Wz + bz) * (X·Wh + bh)    -> d_out
// Block = 256 threads (8 waves). Block tile: 512 rows x 32 cols.
// Wave w: 4 M-subtiles at w*64 + {0,16,32,48}, 2 N-subtiles, both matrices
// => 16 accumulators, 16 WMMAs per 16 b128 loads per K-step (~32 FLOP/B).
// ---------------------------------------------------------------------------
__global__ __launch_bounds__(256)
void gemm_gates_kernel(const unsigned short* __restrict__ Xb,
                       const unsigned short* __restrict__ Wzb,
                       const unsigned short* __restrict__ Whb,
                       const float* __restrict__ bz,
                       const float* __restrict__ bh,
                       float* __restrict__ coeff, float* __restrict__ val) {
  const int wave = threadIdx.x >> 5;
  const int lane = threadIdx.x & 31;
  const int half = lane >> 4;        // 0: lanes 0-15, 1: lanes 16-31
  const int lr   = lane & 15;

  const int m0 = blockIdx.x * 512 + wave * 64;   // 4 M subtiles: m0 + 16*ms
  const int n0 = blockIdx.y * 32;                // 2 N subtiles: n0, n0+16

  const unsigned short* arow[4];
  #pragma unroll
  for (int ms = 0; ms < 4; ms++)
    arow[ms] = Xb + (size_t)(m0 + ms * 16 + lr) * KK_DIM;

  const unsigned short* bzrow0 = Wzb + (size_t)(n0 + lr)      * KK_DIM;
  const unsigned short* bzrow1 = Wzb + (size_t)(n0 + 16 + lr) * KK_DIM;
  const unsigned short* bhrow0 = Whb + (size_t)(n0 + lr)      * KK_DIM;
  const unsigned short* bhrow1 = Whb + (size_t)(n0 + 16 + lr) * KK_DIM;

  v8f accz[4][2] = {};   // [msub][nsub], z-gate logits
  v8f acch[4][2] = {};   // [msub][nsub], h_tilde logits

  for (int kk = 0; kk < KK_DIM; kk += 32) {
    v16bf wz0 = load_b_frag(bzrow0, kk, half);
    v16bf wz1 = load_b_frag(bzrow1, kk, half);
    v16bf wh0 = load_b_frag(bhrow0, kk, half);
    v16bf wh1 = load_b_frag(bhrow1, kk, half);
    #pragma unroll
    for (int ms = 0; ms < 4; ms++) {
      v16bf a = load_a_frag(arow[ms], kk, half);
      accz[ms][0] = wmma_bf16(a, wz0, accz[ms][0]);
      accz[ms][1] = wmma_bf16(a, wz1, accz[ms][1]);
      acch[ms][0] = wmma_bf16(a, wh0, acch[ms][0]);
      acch[ms][1] = wmma_bf16(a, wh1, acch[ms][1]);
    }
  }

  // Epilogue. C/D layout: element r of v8f, lane L -> M = r + 8*(L>=16), N = L%16.
  #pragma unroll
  for (int ms = 0; ms < 4; ms++) {
    #pragma unroll
    for (int ns = 0; ns < 2; ns++) {
      const int col = n0 + ns * 16 + lr;
      const float bzv = bz[col];
      const float bhv = bh[col];
      const int rowbase = m0 + ms * 16 + half * 8;
      #pragma unroll
      for (int r = 0; r < 8; r++) {
        const size_t idx = (size_t)(rowbase + r) * HH + col;
        const float gz = accz[ms][ns][r] + bzv;
        const float z  = 1.0f / (1.0f + __expf(-gz));
        const float ht = acch[ms][ns][r] + bhv;
        coeff[idx] = 1.0f - z;
        val[idx]   = z * ht;
      }
    }
  }
}

// ---------------------------------------------------------------------------
// Scan phase 1: per-chunk affine aggregate.  h_end = A * h_start + S
// grid = (HH/256, BB*NCHUNK), block = 256. Lane -> h (coalesced over h).
// ---------------------------------------------------------------------------
__global__ __launch_bounds__(256)
void scan_chunks_kernel(const float* __restrict__ coeff,
                        const float* __restrict__ val,
                        float* __restrict__ Ach, float* __restrict__ Sch) {
  const int h  = blockIdx.x * blockDim.x + threadIdx.x;
  const int bc = blockIdx.y;
  const int b  = bc >> 5;            // NCHUNK == 32
  const int c  = bc & 31;
  size_t base = ((size_t)(b * TT + c * CLEN)) * HH + h;
  float A = 1.0f, S = 0.0f;
  for (int t = 0; t < CLEN; t++) {
    const float a = coeff[base];
    const float v = val[base];
    S = fmaf(a, S, v);
    A *= a;
    base += HH;
  }
  const size_t idx = (size_t)bc * HH + h;
  Ach[idx] = A;
  Sch[idx] = S;
}

// ---------------------------------------------------------------------------
// Scan phase 2: sequential carry over the 32 chunks, seeded with h_0.
// One thread per (b, h): 8192 threads.
// ---------------------------------------------------------------------------
__global__ __launch_bounds__(256)
void scan_carry_kernel(const float* __restrict__ h0,
                       const float* __restrict__ Ach,
                       const float* __restrict__ Sch,
                       float* __restrict__ Hst) {
  const int idx = blockIdx.x * blockDim.x + threadIdx.x;  // b*HH + h
  const int b = idx >> 10;
  const int h = idx & 1023;
  float hr = h0[idx];
  for (int c = 0; c < NCHUNK; c++) {
    const size_t i = (size_t)(b * NCHUNK + c) * HH + h;
    Hst[i] = hr;
    hr = fmaf(Ach[i], hr, Sch[i]);
  }
}

// ---------------------------------------------------------------------------
// Scan phase 3: apply. Reads coeff (ws) + val (d_out), writes h into d_out
// in place (read-before-write per element within the same thread).
// ---------------------------------------------------------------------------
__global__ __launch_bounds__(256)
void scan_apply_kernel(const float* __restrict__ coeff,
                       const float* __restrict__ Hst,
                       float* out) {
  const int h  = blockIdx.x * blockDim.x + threadIdx.x;
  const int bc = blockIdx.y;
  const int b  = bc >> 5;
  const int c  = bc & 31;
  float hr = Hst[(size_t)bc * HH + h];
  size_t base = ((size_t)(b * TT + c * CLEN)) * HH + h;
  for (int t = 0; t < CLEN; t++) {
    const float a = coeff[base];
    const float v = out[base];
    hr = fmaf(a, hr, v);
    out[base] = hr;
    base += HH;
  }
}

// ---------------------------------------------------------------------------
// Launch
// ---------------------------------------------------------------------------
extern "C" void kernel_launch(void* const* d_in, const int* in_sizes, int n_in,
                              void* d_out, int out_size, void* d_ws, size_t ws_size,
                              hipStream_t stream) {
  const float* x  = (const float*)d_in[0];  // (B, T, DX)
  const float* h0 = (const float*)d_in[1];  // (B, 1, DH)
  const float* Wz = (const float*)d_in[2];  // (DH, DX)
  const float* bz = (const float*)d_in[3];  // (DH,)
  const float* Wh = (const float*)d_in[4];  // (DH, DX)
  const float* bh = (const float*)d_in[5];  // (DH,)
  float* out = (float*)d_out;               // (B, T, DH); also holds `val`

  const size_t NXT = (size_t)BB * TT * HH;      // 33,554,432
  const size_t NW  = (size_t)HH * KK_DIM;       // 1,048,576
  const size_t NAG = (size_t)BB * NCHUNK * HH;  // 262,144

  // Workspace layout: coeff(f32) | Ach | Sch | Hst | Xbf(bf16) | Wzbf | Whbf
  float* coeff = (float*)d_ws;
  float* Ach   = coeff + NXT;
  float* Sch   = Ach + NAG;
  float* Hst   = Sch + NAG;
  unsigned short* Xbf  = (unsigned short*)(Hst + NAG);
  unsigned short* Wzbf = Xbf + NXT;
  unsigned short* Whbf = Wzbf + NW;

  // 0) f32 -> bf16 pre-pass (memory bound; hides conversion VALU)
  cvt_bf16_kernel<<<(int)(NXT / 4 / 256), 256, 0, stream>>>(
      x, (unsigned int*)Xbf, (int)(NXT / 4));
  cvt_bf16_kernel<<<(int)(NW / 4 / 256), 256, 0, stream>>>(
      Wz, (unsigned int*)Wzbf, (int)(NW / 4));
  cvt_bf16_kernel<<<(int)(NW / 4 / 256), 256, 0, stream>>>(
      Wh, (unsigned int*)Whbf, (int)(NW / 4));

  // 1) dual GEMM + gates: grid (M/512, H/32)
  dim3 ggrid((BB * TT) / 512, HH / 32);
  gemm_gates_kernel<<<ggrid, 256, 0, stream>>>(Xbf, Wzbf, Whbf, bz, bh,
                                               coeff, out);

  // 2) chunk aggregates
  dim3 sgrid(HH / 256, BB * NCHUNK);
  scan_chunks_kernel<<<sgrid, 256, 0, stream>>>(coeff, out, Ach, Sch);

  // 3) carry scan over chunks
  scan_carry_kernel<<<(BB * HH) / 256, 256, 0, stream>>>(h0, Ach, Sch, Hst);

  // 4) apply within chunks, in-place into d_out
  scan_apply_kernel<<<sgrid, 256, 0, stream>>>(coeff, Hst, out);
}
